// GaussianKernel_67370857005321
// MI455X (gfx1250) — matrix operand order repaired and checked by reading, not compiled, not run on hardware
//
#include <hip/hip_runtime.h>

#define N_ROWS 4096
#define K_DIM  1024

typedef __attribute__((ext_vector_type(16))) __bf16        v16bf;
typedef __attribute__((ext_vector_type(8)))  float         v8f;
typedef __attribute__((ext_vector_type(4)))  unsigned int  v4u;

union FragU { v4u u[2]; v16bf f; };

// ---- bf16 helpers (bit tricks, no scalar __bf16 arithmetic) ----
__device__ __forceinline__ unsigned short f32_to_bf16_rne(float f) {
    unsigned int u = __builtin_bit_cast(unsigned int, f);
    unsigned int r = u + 0x7FFFu + ((u >> 16) & 1u);
    return (unsigned short)(r >> 16);
}
__device__ __forceinline__ float bf16_to_f32(unsigned short h) {
    unsigned int u = ((unsigned int)h) << 16;
    return __builtin_bit_cast(float, u);
}

// ======================================================================
// Kernel 1: split X (fp32) into bf16 hi/lo and compute fp32 row norms
// grid = 4096 blocks (one per row), 256 threads (4 cols each)
// ======================================================================
__global__ __launch_bounds__(256)
void gk_prep(const float* __restrict__ X,
             unsigned short* __restrict__ Xhi,
             unsigned short* __restrict__ Xlo,
             float* __restrict__ sq) {
    const int row = blockIdx.x;
    const int t   = threadIdx.x;
    const size_t base = (size_t)row * K_DIM + (size_t)t * 4;

    float4 v = *(const float4*)(X + base);
    float vv[4] = { v.x, v.y, v.z, v.w };

    unsigned long long phi = 0ull, plo = 0ull;
    float s = 0.0f;
#pragma unroll
    for (int i = 0; i < 4; ++i) {
        float x = vv[i];
        s += x * x;
        unsigned short hi = f32_to_bf16_rne(x);
        float residual = x - bf16_to_f32(hi);
        unsigned short lo = f32_to_bf16_rne(residual);
        phi |= ((unsigned long long)hi) << (16 * i);
        plo |= ((unsigned long long)lo) << (16 * i);
    }
    *(unsigned long long*)(Xhi + base) = phi;
    *(unsigned long long*)(Xlo + base) = plo;

    __shared__ float red[256];
    red[t] = s;
    __syncthreads();
#pragma unroll
    for (int off = 128; off > 0; off >>= 1) {
        if (t < off) red[t] += red[t + off];
        __syncthreads();
    }
    if (t == 0) sq[row] = red[0];
}

// ======================================================================
// Kernel 2: 128x128 tile WMMA GEMM (hi/lo split) + Gaussian epilogue
// 256 threads = 8 wave32s in a 2(M) x 4(N) grid; each wave: 64x32 out
// LDS: 4 tiles (Ahi,Alo,Bhi,Blo), 128 rows x 32 bf16, row stride 80B
// ======================================================================
__global__ __launch_bounds__(256)
void gk_gemm(const unsigned short* __restrict__ Xhi,
             const unsigned short* __restrict__ Xlo,
             const float* __restrict__ sq,
             float* __restrict__ out) {
    // 4 tiles * 128 rows * 5 v4u (64B data + 16B pad) = 40960 B
    __shared__ v4u lds[4 * 128 * 5];

    const int tid  = threadIdx.x;
    const int lane = tid & 31;
    const int w    = tid >> 5;
    const int wm   = w >> 2;        // 0..1  (M direction, 64 rows each)
    const int wn   = w & 3;         // 0..3  (N direction, 32 cols each)
    const int mb   = blockIdx.y * 128;
    const int nb   = blockIdx.x * 128;

    const int lrow  = lane & 15;
    const int lhalf = lane >> 4;    // 0 or 1

    // A-fragment chunk offsets (v4u units within a tile):
    //   lane<16: K0-7 @ +0, K16-23 @ +32B ; lane>=16: K8-15 @ +16B, K24-31 @ +48B
    const int fa0 = lrow * 5 + lhalf;       // * 16B
    const int fa1 = fa0 + 2;
    // B-fragment: same as A: lane<16 K0-7,K16-23 ; lane>=16 K8-15,K24-31
    const int fb0 = lrow * 5 + lhalf;
    const int fb1 = fb0 + 2;

    v8f zero = {};
    v8f acc[4][2];
#pragma unroll
    for (int i = 0; i < 4; ++i)
#pragma unroll
        for (int j = 0; j < 2; ++j) acc[i][j] = zero;

    for (int kt = 0; kt < K_DIM; kt += 32) {
        // ---- stage 4 tiles: global -> LDS (b128 per thread, 2 passes/tile) ----
#pragma unroll
        for (int tile = 0; tile < 4; ++tile) {
            const unsigned short* src = (tile & 1) ? Xlo : Xhi;
            const int rb = (tile < 2) ? mb : nb;
#pragma unroll
            for (int p = 0; p < 2; ++p) {
                int idx = tid + p * 256;         // 0..511
                int row = idx >> 2;              // 0..127
                int c   = idx & 3;               // 16B chunk in row
                v4u d = *(const v4u*)(src + (size_t)(rb + row) * K_DIM + kt + c * 8);
                lds[tile * 640 + row * 5 + c] = d;
            }
        }
        __syncthreads();

        // ---- load fragments from LDS ----
        FragU ahi[4], alo[4], bhi[2], blo[2];
#pragma unroll
        for (int i = 0; i < 4; ++i) {
            int r = (wm * 64 + i * 16) * 5;
            ahi[i].u[0] = lds[0 * 640 + r + fa0];
            ahi[i].u[1] = lds[0 * 640 + r + fa1];
            alo[i].u[0] = lds[1 * 640 + r + fa0];
            alo[i].u[1] = lds[1 * 640 + r + fa1];
        }
#pragma unroll
        for (int j = 0; j < 2; ++j) {
            int r = (wn * 32 + j * 16) * 5;
            bhi[j].u[0] = lds[2 * 640 + r + fb0];
            bhi[j].u[1] = lds[2 * 640 + r + fb1];
            blo[j].u[0] = lds[3 * 640 + r + fb0];
            blo[j].u[1] = lds[3 * 640 + r + fb1];
        }

        // ---- 24 WMMAs: hi*hi + hi*lo + lo*hi ----
#pragma unroll
        for (int i = 0; i < 4; ++i) {
#pragma unroll
            for (int j = 0; j < 2; ++j) {
                v8f c = acc[i][j];
                c = __builtin_amdgcn_wmma_f32_16x16x32_bf16(
                        false, ahi[i].f, false, bhi[j].f, (short)0, c, false, false);
                c = __builtin_amdgcn_wmma_f32_16x16x32_bf16(
                        false, ahi[i].f, false, blo[j].f, (short)0, c, false, false);
                c = __builtin_amdgcn_wmma_f32_16x16x32_bf16(
                        false, alo[i].f, false, bhi[j].f, (short)0, c, false, false);
                acc[i][j] = c;
            }
        }
        __syncthreads();
    }

    // ---- epilogue: d2 = sq[m] + sq[n] - 2*gram ; out = exp(-d2/2) ----
#pragma unroll
    for (int i = 0; i < 4; ++i) {
#pragma unroll
        for (int j = 0; j < 2; ++j) {
            int n = nb + wn * 32 + j * 16 + lrow;
            float sn = sq[n];
#pragma unroll
            for (int r = 0; r < 8; ++r) {
                int m = mb + wm * 64 + i * 16 + r + lhalf * 8;
                float g  = acc[i][j][r];
                float d2 = sq[m] + sn - 2.0f * g;
                d2 = fmaxf(d2, 0.0f);
                out[(size_t)m * N_ROWS + n] = __expf(-0.5f * d2);
            }
        }
    }
}

// ======================================================================
extern "C" void kernel_launch(void* const* d_in, const int* in_sizes, int n_in,
                              void* d_out, int out_size, void* d_ws, size_t ws_size,
                              hipStream_t stream) {
    const float* X = (const float*)d_in[0];

    unsigned short* Xhi = (unsigned short*)d_ws;                    // 8 MB
    unsigned short* Xlo = Xhi + (size_t)N_ROWS * K_DIM;             // 8 MB
    float*          sq  = (float*)(Xlo + (size_t)N_ROWS * K_DIM);   // 16 KB

    gk_prep<<<N_ROWS, 256, 0, stream>>>(X, Xhi, Xlo, sq);

    dim3 grid(N_ROWS / 128, N_ROWS / 128, 1);   // 32 x 32 tiles
    gk_gemm<<<grid, 256, 0, stream>>>(Xhi, Xlo, sq, (float*)d_out);
}
